// HeavilyCompressedAttention_10582799417771
// MI455X (gfx1250) — matrix operand order repaired and checked by reading, not compiled, not run on hardware
//
#include <hip/hip_runtime.h>
#include <hip/hip_bf16.h>

#define TT 8192
#define DD 256
#define CC 64
#define LKV 528
#define LPAD 544

typedef __attribute__((ext_vector_type(16))) __bf16 v16bf;
typedef __attribute__((ext_vector_type(8)))  float  v8f;

static __device__ __forceinline__ v8f wmma_bf16(v16bf a, v16bf b, v8f c) {
  // D = A(16x32 bf16) x B(32x16 bf16) + C(16x16 f32)
  return __builtin_amdgcn_wmma_f32_16x16x32_bf16(false, a, false, b, (short)0, c, false, false);
}

static __device__ __forceinline__ v8f zero8() {
  v8f z = {0.f, 0.f, 0.f, 0.f, 0.f, 0.f, 0.f, 0.f};
  return z;
}

// A-matrix fragment: lane holds row m=lane%16, elements 0..7 = k (hi*8 .. hi*8+7),
// elements 8..15 = k (16+hi*8 .. 16+hi*8+7)  -> two contiguous 16B chunks.
static __device__ __forceinline__ v16bf load_a(const __hip_bfloat16* p0,
                                               const __hip_bfloat16* p1) {
  union { uint4 q[2]; v16bf v; } u;
  u.q[0] = *reinterpret_cast<const uint4*>(p0);
  u.q[1] = *reinterpret_cast<const uint4*>(p1);
  return u.v;
}

// B-matrix fragment: lane holds column n=lane%16, elements 0..7 = k (hi*8 .. hi*8+7),
// elements 8..15 = k (16+hi*8 .. 16+hi*8+7) -> two 16B chunks (p points at k=hi*16).
static __device__ __forceinline__ v16bf load_b(const __hip_bfloat16* p) {
  union { uint4 q[2]; v16bf v; } u;
  const int hb = (threadIdx.x >> 4) & 1;
  const __hip_bfloat16* base = p - hb * 16;
  u.q[0] = *reinterpret_cast<const uint4*>(base + hb * 8);
  u.q[1] = *reinterpret_cast<const uint4*>(base + 16 + hb * 8);
  return u.v;
}

// reductions within a 16-lane half (row of C-layout lives in one half of the wave32)
static __device__ __forceinline__ float red_max16(float v) {
  v = fmaxf(v, __shfl_xor(v, 1, 32));
  v = fmaxf(v, __shfl_xor(v, 2, 32));
  v = fmaxf(v, __shfl_xor(v, 4, 32));
  v = fmaxf(v, __shfl_xor(v, 8, 32));
  return v;
}
static __device__ __forceinline__ float red_sum16(float v) {
  v += __shfl_xor(v, 1, 32);
  v += __shfl_xor(v, 2, 32);
  v += __shfl_xor(v, 4, 32);
  v += __shfl_xor(v, 8, 32);
  return v;
}

// CDNA5 async global->LDS copy: 16 bytes per lane, tracked by ASYNCcnt.
// Low 32 bits of a flat LDS pointer are the wave-relative LDS byte address.
static __device__ __forceinline__ void async_copy16(unsigned lds_off, const void* gaddr) {
  asm volatile("global_load_async_to_lds_b128 %0, %1, off"
               :: "v"(lds_off), "v"((unsigned long long)(uintptr_t)gaddr)
               : "memory");
}
static __device__ __forceinline__ void wait_async0() {
  asm volatile("s_wait_asynccnt 0x0" ::: "memory");
}
// Wave-private LDS RAW/WAR ordering: DS ops complete in order within a wave,
// we only need to stop the compiler from reordering around the staging buffer.
static __device__ __forceinline__ void lds_fence() {
  asm volatile("" ::: "memory");
}

// ---------------------------------------------------------------------------
// Prep 1: transpose + convert all GEMM B-operands to bf16.
// Wqt[c][d]   = W_Q[d][c]     (256 x 256)
// Wg0t[c][k]  = Wg0[k][c]     (64 x 128), same for Wg1t
// Woutt[c][k] = Wout[k][c]    (256 x 128)
// ---------------------------------------------------------------------------
__global__ void prep_weights(const float* __restrict__ WQ,
                             const float* __restrict__ Wg0,
                             const float* __restrict__ Wg1,
                             const float* __restrict__ Wout,
                             __hip_bfloat16* __restrict__ Wqt,
                             __hip_bfloat16* __restrict__ Wg0t,
                             __hip_bfloat16* __restrict__ Wg1t,
                             __hip_bfloat16* __restrict__ Woutt) {
  int i = blockIdx.x * blockDim.x + threadIdx.x;
  if (i < 256 * 256) {
    int c = i >> 8, d = i & 255;
    Wqt[i] = __float2bfloat16(WQ[d * 256 + c]);
  }
  if (i < 64 * 128) {
    int c = i >> 7, k = i & 127;
    Wg0t[i] = __float2bfloat16(Wg0[k * 64 + c]);
    Wg1t[i] = __float2bfloat16(Wg1[k * 64 + c]);
  }
  if (i < 256 * 128) {
    int c = i >> 7, k = i & 127;
    Woutt[i] = __float2bfloat16(Wout[k * 256 + c]);
  }
}

// ---------------------------------------------------------------------------
// Prep 2: KV path. One block per (l, b); 64 threads (one per channel).
// l < 512: compressed token (mean of 16 H rows) @ W_KV;  l in [512,528): sliding.
// RMSNorm (w_k / w_v), RoPE at position l.  K stored [b][l][c] (row-major,
// padded to 544 rows with zeros); V stored transposed [b][c][l] so that the
// PV GEMM's B fragments are contiguous.
// ---------------------------------------------------------------------------
__global__ __launch_bounds__(64) void prep_kv(const float* __restrict__ H,
                                              const float* __restrict__ WKV,
                                              const float* __restrict__ wk,
                                              const float* __restrict__ wv,
                                              __hip_bfloat16* __restrict__ Kbf,
                                              __hip_bfloat16* __restrict__ Vtbf) {
  const int l = blockIdx.x, b = blockIdx.y, c = threadIdx.x;
  __shared__ float h[256];
  __shared__ float kn[64], vn[64], red[64];

  if (l >= LKV) {  // zero padding rows (uniform branch)
    Kbf[((size_t)b * LPAD + l) * 64 + c] = __float2bfloat16(0.f);
    Vtbf[(size_t)b * 64 * LPAD + (size_t)c * LPAD + l] = __float2bfloat16(0.f);
    return;
  }

  if (l < 512) {
    for (int d = c; d < 256; d += 64) {
      const float* base = H + ((size_t)b * TT + (size_t)l * 16) * 256 + d;
      float s = 0.f;
#pragma unroll
      for (int i = 0; i < 16; ++i) s += base[(size_t)i * 256];
      h[d] = s * (1.f / 16.f);
    }
  } else {
    int row = TT - 16 + (l - 512);
    for (int d = c; d < 256; d += 64) h[d] = H[((size_t)b * TT + row) * 256 + d];
  }
  __syncthreads();

  float kv = 0.f;
  for (int d = 0; d < 256; ++d) kv += h[d] * WKV[d * 64 + c];

  red[c] = kv * kv;
  __syncthreads();
  for (int s = 32; s > 0; s >>= 1) {
    if (c < s) red[c] += red[c + s];
    __syncthreads();
  }
  float rms = rsqrtf(red[0] * (1.f / 64.f) + 1e-6f);
  kn[c] = kv * rms * wk[c];
  vn[c] = kv * rms * wv[c];
  __syncthreads();

  int f = c & 31;
  float invf = __powf(10000.f, -(float)f * (1.f / 32.f));
  float sn, cs;
  __sincosf((float)l * invf, &sn, &cs);
  int pr = c ^ 32;
  float kr, vr;
  if (c < 32) { kr = kn[c] * cs - kn[pr] * sn; vr = vn[c] * cs - vn[pr] * sn; }
  else        { kr = kn[c] * cs + kn[pr] * sn; vr = vn[c] * cs + vn[pr] * sn; }
  Kbf[((size_t)b * LPAD + l) * 64 + c] = __float2bfloat16(kr);
  Vtbf[(size_t)b * 64 * LPAD + (size_t)c * LPAD + l] = __float2bfloat16(vr);
}

// ---------------------------------------------------------------------------
// Attention: one block = 16 query rows, 4 waves = 4 heads.
// The block's full (padded) K and V^T for this batch (2 x 68 KB) are staged
// into LDS with async global->LDS copies, overlapped with the Q-projection
// WMMAs; the flash-softmax loop then feeds all B fragments from LDS.
// Dynamic LDS carve (156 KB, fits CDNA5's 320 KB/WGP):
//   Ks [544][64] bf16, Vs [64][544] bf16, Hs [16][256], Qs [4][16][64],
//   Ps [4][16][32]
// ---------------------------------------------------------------------------
#define SM_KS 0
#define SM_VS 69632
#define SM_HS 139264
#define SM_QS 147456
#define SM_PS 155648
#define SM_TOTAL 159744

__global__ __launch_bounds__(128) void attn_kernel(const float* __restrict__ H,
                                                   const float* __restrict__ wq,
                                                   const __hip_bfloat16* __restrict__ Wqt,
                                                   const __hip_bfloat16* __restrict__ Kbf,
                                                   const __hip_bfloat16* __restrict__ Vtbf,
                                                   __hip_bfloat16* __restrict__ Obf) {
  extern __shared__ __align__(16) char smem[];
  __hip_bfloat16* Ks = (__hip_bfloat16*)(smem + SM_KS);   // [l*64 + c]
  __hip_bfloat16* Vs = (__hip_bfloat16*)(smem + SM_VS);   // [c*544 + l]
  __hip_bfloat16* Hs = (__hip_bfloat16*)(smem + SM_HS);   // [m*256 + d]
  __hip_bfloat16* Qs = (__hip_bfloat16*)(smem + SM_QS);   // [(head*16+m)*64 + c]
  __hip_bfloat16* Ps = (__hip_bfloat16*)(smem + SM_PS);   // [(head*16+m)*32 + k]

  const int tile = blockIdx.x, b = blockIdx.y;
  const int tid = threadIdx.x;
  const int head = tid >> 5;
  const int lane = tid & 31;
  const int n = lane & 15;
  const int hi = lane >> 4;

  // ---- kick off async K/V staging (overlaps everything up to the attn loop)
  const __hip_bfloat16* Kb = Kbf + (size_t)b * LPAD * CC;
  const __hip_bfloat16* Vt = Vtbf + (size_t)b * CC * LPAD;
  {
    const unsigned ldsK = (unsigned)(uintptr_t)Ks;
    const unsigned ldsV = (unsigned)(uintptr_t)Vs;
    const int elt = lane * 8;                      // 8 bf16 = 16 B per lane
    for (int ch = head; ch < 136; ch += 4) {       // 136 chunks x 512 B = 68 KB
      int eoff = ch * 256 + elt;
      async_copy16(ldsK + eoff * 2, Kb + eoff);
      async_copy16(ldsV + eoff * 2, Vt + eoff);
    }
  }

  // ---- cooperative H tile load (fp32 -> bf16 in LDS)
  const float* Hrow = H + ((size_t)b * TT + (size_t)tile * 16) * DD;
  for (int i = tid; i < 16 * DD; i += 128)
    Hs[i] = __float2bfloat16(Hrow[i]);
  if (tile + 1 < TT / 16) __builtin_prefetch(Hrow + 16 * DD, 0, 1);
  __syncthreads();

  // ---- Q projection: 4 n-tiles x 8 k-steps of WMMA ----
  v8f qacc[4];
#pragma unroll
  for (int nt = 0; nt < 4; ++nt) qacc[nt] = zero8();
#pragma unroll
  for (int ks = 0; ks < 8; ++ks) {
    v16bf a = load_a(&Hs[n * 256 + ks * 32 + hi * 8],
                     &Hs[n * 256 + ks * 32 + 16 + hi * 8]);
#pragma unroll
    for (int nt = 0; nt < 4; ++nt) {
      const __hip_bfloat16* br =
          Wqt + (size_t)(head * 64 + nt * 16 + n) * 256 + ks * 32 + hi * 16;
      qacc[nt] = wmma_bf16(a, load_b(br), qacc[nt]);
    }
  }

  float q[4][8];
#pragma unroll
  for (int nt = 0; nt < 4; ++nt)
#pragma unroll
    for (int r = 0; r < 8; ++r) q[nt][r] = qacc[nt][r];

  // RMSNorm over C=64 (4 local channels + 16 lanes) and w_q scale
  float wqv[4];
#pragma unroll
  for (int nt = 0; nt < 4; ++nt) wqv[nt] = wq[head * 64 + nt * 16 + n];
#pragma unroll
  for (int r = 0; r < 8; ++r) {
    float s = q[0][r] * q[0][r] + q[1][r] * q[1][r] + q[2][r] * q[2][r] + q[3][r] * q[3][r];
    s = red_sum16(s);
    float rms = rsqrtf(s * (1.f / 64.f) + 1e-6f);
#pragma unroll
    for (int nt = 0; nt < 4; ++nt) q[nt][r] *= rms * wqv[nt];
  }

  // RoPE(Q): channel c = nt*16+n; partner channel (c +/- 32) is nt^2, in-register
  const float invf0 = __powf(10000.f, -(float)n * (1.f / 32.f));
  const float invf1 = __powf(10000.f, -(float)(16 + n) * (1.f / 32.f));
  __hip_bfloat16* Qh = Qs + head * 16 * 64;
#pragma unroll
  for (int r = 0; r < 8; ++r) {
    float pos = (float)(tile * 16 + hi * 8 + r);
    float s0, c0, s1, c1;
    __sincosf(pos * invf0, &s0, &c0);
    __sincosf(pos * invf1, &s1, &c1);
    float o0 = q[0][r] * c0 - q[2][r] * s0;
    float o1 = q[1][r] * c1 - q[3][r] * s1;
    float o2 = q[2][r] * c0 + q[0][r] * s0;
    float o3 = q[3][r] * c1 + q[1][r] * s1;
    int m = hi * 8 + r;
    Qh[m * 64 + 0 + n]  = __float2bfloat16(o0);
    Qh[m * 64 + 16 + n] = __float2bfloat16(o1);
    Qh[m * 64 + 32 + n] = __float2bfloat16(o2);
    Qh[m * 64 + 48 + n] = __float2bfloat16(o3);
  }
  lds_fence();          // Qs is wave-private; HW keeps DS ops in order per wave

  // make the async K/V staging visible to all waves
  wait_async0();
  __syncthreads();

  // ---- online-softmax attention over 544 (padded) KV rows, all from LDS ----
  v8f oacc[4];
#pragma unroll
  for (int nt = 0; nt < 4; ++nt) oacc[nt] = zero8();
  float mmax[8], msum[8];
#pragma unroll
  for (int r = 0; r < 8; ++r) { mmax[r] = -1e30f; msum[r] = 0.f; }

  __hip_bfloat16* Ph = Ps + head * 16 * 32;

  for (int it = 0; it < LPAD / 32; ++it) {
    // S chunk: 16 x 32 = two 16x16 tiles, K-dim = channel (2 x 32)
    v8f sa0 = zero8(), sa1 = zero8();
#pragma unroll
    for (int ks = 0; ks < 2; ++ks) {
      v16bf a = load_a(&Qh[n * 64 + ks * 32 + hi * 8],
                       &Qh[n * 64 + ks * 32 + 16 + hi * 8]);
      sa0 = wmma_bf16(a, load_b(&Ks[(it * 32 + n) * 64 + ks * 32 + hi * 16]), sa0);
      sa1 = wmma_bf16(a, load_b(&Ks[(it * 32 + 16 + n) * 64 + ks * 32 + hi * 16]), sa1);
    }
    const int col0 = it * 32 + n;
    const int col1 = col0 + 16;
    float p0[8], p1[8];
#pragma unroll
    for (int r = 0; r < 8; ++r) {
      float s0 = sa0[r] * 0.125f;           // 1/sqrt(64)
      float s1 = sa1[r] * 0.125f;
      float rmax = red_max16(fmaxf(s0, s1));
      rmax = fmaxf(rmax, mmax[r]);
      float fac = __expf(mmax[r] - rmax);
      mmax[r] = rmax;
      p0[r] = (col0 < LKV) ? __expf(s0 - rmax) : 0.f;
      p1[r] = (col1 < LKV) ? __expf(s1 - rmax) : 0.f;
      msum[r] = msum[r] * fac + red_sum16(p0[r] + p1[r]);
#pragma unroll
      for (int nt = 0; nt < 4; ++nt) oacc[nt][r] *= fac;
    }
    // stage P (16x32 bf16) through wave-private LDS: C-layout -> A-layout
#pragma unroll
    for (int r = 0; r < 8; ++r) {
      int m = hi * 8 + r;
      Ph[m * 32 + n]      = __float2bfloat16(p0[r]);
      Ph[m * 32 + 16 + n] = __float2bfloat16(p1[r]);
    }
    lds_fence();
    v16bf pa = load_a(&Ph[n * 32 + hi * 8], &Ph[n * 32 + 16 + hi * 8]);
#pragma unroll
    for (int nt = 0; nt < 4; ++nt) {
      oacc[nt] = wmma_bf16(pa, load_b(&Vs[(nt * 16 + n) * LPAD + it * 32 + hi * 16]),
                           oacc[nt]);
    }
    lds_fence();
  }

  // normalize, inverse RoPE, store bf16
  float o[4][8];
#pragma unroll
  for (int nt = 0; nt < 4; ++nt)
#pragma unroll
    for (int r = 0; r < 8; ++r) o[nt][r] = oacc[nt][r] / msum[r];

  __hip_bfloat16* Ob = Obf + ((size_t)b * TT + (size_t)tile * 16) * 256 + head * 64;
#pragma unroll
  for (int r = 0; r < 8; ++r) {
    float pos = (float)(tile * 16 + hi * 8 + r);
    float s0, c0, s1, c1;
    __sincosf(pos * invf0, &s0, &c0);
    __sincosf(pos * invf1, &s1, &c1);
    float r0 = o[0][r] * c0 + o[2][r] * s0;   // negate=True rope
    float r1 = o[1][r] * c1 + o[3][r] * s1;
    float r2 = o[2][r] * c0 - o[0][r] * s0;
    float r3 = o[3][r] * c1 - o[1][r] * s1;
    __hip_bfloat16* row = Ob + (size_t)(hi * 8 + r) * 256;
    row[0 + n]  = __float2bfloat16(r0);
    row[16 + n] = __float2bfloat16(r1);
    row[32 + n] = __float2bfloat16(r2);
    row[48 + n] = __float2bfloat16(r3);
  }
}

// ---------------------------------------------------------------------------
// Output MLP: p_all = [Og0@Wg0+bg0 , Og1@Wg1+bg1] (16x128), out = p_all@Wout+bout.
// 4 waves split the 8 p-tiles (2 each) then the 16 out-tiles (4 each).
// ---------------------------------------------------------------------------
__global__ __launch_bounds__(128) void mlp_kernel(const __hip_bfloat16* __restrict__ Obf,
                                                  const __hip_bfloat16* __restrict__ Wg0t,
                                                  const __hip_bfloat16* __restrict__ Wg1t,
                                                  const __hip_bfloat16* __restrict__ Woutt,
                                                  const float* __restrict__ bg0,
                                                  const float* __restrict__ bg1,
                                                  const float* __restrict__ bout,
                                                  float* __restrict__ out) {
  __shared__ __align__(16) __hip_bfloat16 Os[16][256];
  __shared__ __align__(16) __hip_bfloat16 Pall[16][128];

  const int tile = blockIdx.x, b = blockIdx.y;
  const int tid = threadIdx.x;
  const int wave = tid >> 5;
  const int lane = tid & 31;
  const int n = lane & 15;
  const int hi = lane >> 4;

  const __hip_bfloat16* Orow = Obf + ((size_t)b * TT + (size_t)tile * 16) * 256;
  for (int i = tid; i < 16 * 256; i += 128) Os[i >> 8][i & 255] = Orow[i];
  __syncthreads();

#pragma unroll
  for (int q2 = 0; q2 < 2; ++q2) {
    int ocol = wave * 2 + q2;           // 0..7 column tile of p_all
    int g = ocol >> 2, nt = ocol & 3;   // group 0: heads01@Wg0, group 1: heads23@Wg1
    const __hip_bfloat16* Wt = g ? Wg1t : Wg0t;
    const float* bg = g ? bg1 : bg0;
    v8f acc = zero8();
#pragma unroll
    for (int ks = 0; ks < 4; ++ks) {
      v16bf a = load_a(&Os[n][g * 128 + ks * 32 + hi * 8],
                       &Os[n][g * 128 + ks * 32 + 16 + hi * 8]);
      const __hip_bfloat16* br = Wt + (size_t)(nt * 16 + n) * 128 + ks * 32 + hi * 16;
      acc = wmma_bf16(a, load_b(br), acc);
    }
    float bias = bg[nt * 16 + n];
#pragma unroll
    for (int r = 0; r < 8; ++r)
      Pall[hi * 8 + r][ocol * 16 + n] = __float2bfloat16(acc[r] + bias);
  }
  __syncthreads();

  float* orow = out + ((size_t)b * TT + (size_t)tile * 16) * 256;
#pragma unroll
  for (int q4 = 0; q4 < 4; ++q4) {
    int nt = wave * 4 + q4;             // 0..15 column tile of output
    v8f acc = zero8();
#pragma unroll
    for (int ks = 0; ks < 4; ++ks) {
      v16bf a = load_a(&Pall[n][ks * 32 + hi * 8], &Pall[n][ks * 32 + 16 + hi * 8]);
      const __hip_bfloat16* br = Woutt + (size_t)(nt * 16 + n) * 128 + ks * 32 + hi * 16;
      acc = wmma_bf16(a, load_b(br), acc);
    }
    float bias = bout[nt * 16 + n];
#pragma unroll
    for (int r = 0; r < 8; ++r)
      orow[(size_t)(hi * 8 + r) * 256 + nt * 16 + n] = acc[r] + bias;
  }
}

// ---------------------------------------------------------------------------
extern "C" void kernel_launch(void* const* d_in, const int* in_sizes, int n_in,
                              void* d_out, int out_size, void* d_ws, size_t ws_size,
                              hipStream_t stream) {
  (void)in_sizes; (void)n_in; (void)out_size; (void)ws_size;
  const float* H    = (const float*)d_in[0];
  const float* WQ   = (const float*)d_in[1];
  const float* WKV  = (const float*)d_in[2];
  const float* wq   = (const float*)d_in[3];
  const float* wk   = (const float*)d_in[4];
  const float* wv   = (const float*)d_in[5];
  const float* Wg0  = (const float*)d_in[6];
  const float* bg0  = (const float*)d_in[7];
  const float* Wg1  = (const float*)d_in[8];
  const float* bg1  = (const float*)d_in[9];
  const float* Wout = (const float*)d_in[10];
  const float* bout = (const float*)d_in[11];
  float* out = (float*)d_out;

  char* ws = (char*)d_ws;
  __hip_bfloat16* Wqt   = (__hip_bfloat16*)(ws + 0);        // 131072 B
  __hip_bfloat16* Wg0t  = (__hip_bfloat16*)(ws + 131072);   //  16384 B
  __hip_bfloat16* Wg1t  = (__hip_bfloat16*)(ws + 147456);   //  16384 B
  __hip_bfloat16* Woutt = (__hip_bfloat16*)(ws + 163840);   //  65536 B
  __hip_bfloat16* Kbf   = (__hip_bfloat16*)(ws + 229376);   // 4*544*64*2 = 278528 B
  __hip_bfloat16* Vtbf  = (__hip_bfloat16*)(ws + 507904);   // 278528 B
  __hip_bfloat16* Obf   = (__hip_bfloat16*)(ws + 786432);   // 4*8192*256*2 = 16 MiB

  // allow 156 KB dynamic LDS for the attention kernel (CDNA5: 320 KB / WGP)
  (void)hipFuncSetAttribute((const void*)attn_kernel,
                            hipFuncAttributeMaxDynamicSharedMemorySize, SM_TOTAL);

  prep_weights<<<dim3(256), dim3(256), 0, stream>>>(WQ, Wg0, Wg1, Wout,
                                                    Wqt, Wg0t, Wg1t, Woutt);
  prep_kv<<<dim3(LPAD, 4), dim3(64), 0, stream>>>(H, WKV, wk, wv, Kbf, Vtbf);
  attn_kernel<<<dim3(TT / 16, 4), dim3(128), SM_TOTAL, stream>>>(H, wq, Wqt, Kbf,
                                                                 Vtbf, Obf);
  mlp_kernel<<<dim3(TT / 16, 4), dim3(128), 0, stream>>>(Obf, Wg0t, Wg1t, Woutt,
                                                         bg0, bg1, bout, out);
}
